// Transcoder_65120294142431
// MI455X (gfx1250) — matrix operand order repaired and checked by reading, not compiled, not run on hardware
//
#include <hip/hip_runtime.h>

// ---------------------------------------------------------------------------
// Top-k sparse transcoder forward for MI455X (gfx1250, wave32, WMMA).
//
//   pre   = (mlp_input - encoder_bias) @ enc_w.T + enc_b      [4096, 32768]
//   vals, idx = top_k(pre, 64); vals = relu(vals)
//   features  = scatter(zeros, idx, vals)                     [4096, 32768]
//   pred      = sparse_decode(vals, idx, dec_w)               [4096, 4096]
//   loss      = mean((pred - mlp_output)^2)
//
// d_out layout (float): features | prediction | loss | pred_loss | sparse_loss
// ---------------------------------------------------------------------------

#define T_DIM  4096
#define K_DIM  4096
#define NF     32768
#define KTOP   64

#define BM 256
#define BN 256
#define BK 32
#define LDSS 40   // LDS row stride in halves (80 bytes) -> bank spread, 16B aligned

typedef __attribute__((ext_vector_type(16))) __bf16 v16bf;
typedef __attribute__((ext_vector_type(8)))  __bf16 v8bf;
typedef __attribute__((ext_vector_type(8)))  float  v8f;

static __device__ inline __bf16 f2bf(float f) {
    // round-to-nearest-even fp32 -> bf16
    unsigned u = __builtin_bit_cast(unsigned, f);
    u = (u + 0x7FFFu + ((u >> 16) & 1u)) >> 16;
    unsigned short s = (unsigned short)u;
    return __builtin_bit_cast(__bf16, s);
}

static __device__ inline float4 ld4(const float* p) { return *(const float4*)p; }

static __device__ inline float4 sub4(float4 a, float4 b) {
    float4 r; r.x = a.x - b.x; r.y = a.y - b.y; r.z = a.z - b.z; r.w = a.w - b.w;
    return r;
}

static __device__ inline void cvt_store8(__bf16* dst, float4 x, float4 y) {
    v8bf v;
    v[0] = f2bf(x.x); v[1] = f2bf(x.y); v[2] = f2bf(x.z); v[3] = f2bf(x.w);
    v[4] = f2bf(y.x); v[5] = f2bf(y.y); v[6] = f2bf(y.z); v[7] = f2bf(y.w);
    *(v8bf*)dst = v;   // ds_store_b128
}

// A fragment (16x32, M x K): per ISA table, lane m (0..15) holds K 0-7 then
// 16-23; lane m+16 holds K 8-15 then 24-31. Two ds_load_b128 per lane.
static __device__ inline v16bf load_fragA(const __bf16* base, int lane) {
    int r  = lane & 15;
    int kh = (lane & 16) ? 8 : 0;
    const __bf16* p = base + r * LDSS + kh;
    union { v16bf v; v8bf h[2]; } u;
    u.h[0] = *(const v8bf*)p;
    u.h[1] = *(const v8bf*)(p + 16);
    return u.v;
}

// B fragment (32x16, K x N): lane n (0..15) holds column n, K 0-7 then 16-23;
// lane n+16 holds column n, K 8-15 then 24-31 (same K order as A fragment).
// Column n of B == row n of enc_w, so W rows load exactly like A rows.
static __device__ inline v16bf load_fragB(const __bf16* base, int lane) {
    int r  = lane & 15;
    int kh = (lane & 16) ? 8 : 0;
    const __bf16* p = base + r * LDSS + kh;
    union { v16bf v; v8bf h[2]; } u;
    u.h[0] = *(const v8bf*)p;
    u.h[1] = *(const v8bf*)(p + 16);
    return u.v;
}

// ---------------------------------------------------------------------------
// 1) Encode GEMM: pre = (A - bias) @ W^T + enc_b, bf16 WMMA, fp32 accumulate.
//    512 threads = 16 waves in a 4x4 grid; each wave computes 64x64 of C.
// ---------------------------------------------------------------------------
__global__ __launch_bounds__(512) void encode_gemm_kernel(
    const float* __restrict__ A,     // mlp_input   [T, K]
    const float* __restrict__ W,     // enc_w       [NF, K]
    const float* __restrict__ bias,  // encoder_bias[K]
    const float* __restrict__ encb,  // enc_b       [NF]
    float* __restrict__ pre)         // [T, NF]
{
    __shared__ __bf16 lA[BM * LDSS];
    __shared__ __bf16 lB[BN * LDSS];

    const int tid  = threadIdx.x;
    const int lane = tid & 31;
    const int wave = tid >> 5;
    const int wm   = wave & 3;     // wave row   (0..3) -> 64 rows each
    const int wn   = wave >> 2;    // wave col   (0..3) -> 64 cols each

    const int blockN = blockIdx.x * BN;
    const int blockM = blockIdx.y * BM;

    const int ldRow = tid >> 1;          // 0..255
    const int ldCol = (tid & 1) * 16;    // 0 or 16 (halves of a 32-wide K slab)

    v8f acc[4][4];
#pragma unroll
    for (int i = 0; i < 4; ++i)
#pragma unroll
        for (int j = 0; j < 4; ++j)
            acc[i][j] = (v8f)(0.0f);

    for (int k0 = 0; k0 < K_DIM; k0 += BK) {
        // ---- stage fp32 -> bf16 tiles into LDS (each thread: 16 A + 16 W) ---
        const float* ga = A + (size_t)(blockM + ldRow) * K_DIM + k0 + ldCol;
        const float* gw = W + (size_t)(blockN + ldRow) * K_DIM + k0 + ldCol;
        const float* gb = bias + k0 + ldCol;

        float4 a0 = ld4(ga + 0), a1 = ld4(ga + 4), a2 = ld4(ga + 8), a3 = ld4(ga + 12);
        float4 b0 = ld4(gb + 0), b1 = ld4(gb + 4), b2 = ld4(gb + 8), b3 = ld4(gb + 12);
        a0 = sub4(a0, b0); a1 = sub4(a1, b1); a2 = sub4(a2, b2); a3 = sub4(a3, b3);
        float4 w0 = ld4(gw + 0), w1 = ld4(gw + 4), w2 = ld4(gw + 8), w3 = ld4(gw + 12);

        __bf16* sa = lA + ldRow * LDSS + ldCol;
        __bf16* sb = lB + ldRow * LDSS + ldCol;
        cvt_store8(sa, a0, a1);
        cvt_store8(sa + 8, a2, a3);
        cvt_store8(sb, w0, w1);
        cvt_store8(sb + 8, w2, w3);

        __syncthreads();

        // ---- fragments + 16 WMMAs ------------------------------------------
        v16bf af[4];
#pragma unroll
        for (int i = 0; i < 4; ++i)
            af[i] = load_fragA(lA + (wm * 64 + i * 16) * LDSS, lane);

#pragma unroll
        for (int j = 0; j < 4; ++j) {
            v16bf bfj = load_fragB(lB + (wn * 64 + j * 16) * LDSS, lane);
#pragma unroll
            for (int i = 0; i < 4; ++i) {
                acc[i][j] = __builtin_amdgcn_wmma_f32_16x16x32_bf16(
                    false, af[i], false, bfj, (short)0, acc[i][j], false, false);
            }
        }
        __syncthreads();
    }

    // ---- epilogue: C layout = lane(0-15):M=r,N=lane ; lane(16-31):M=8+r -----
    const int mhalf = (lane & 16) ? 8 : 0;
#pragma unroll
    for (int j = 0; j < 4; ++j) {
        const int col = blockN + wn * 64 + j * 16 + (lane & 15);
        const float bn = encb[col];
#pragma unroll
        for (int i = 0; i < 4; ++i) {
            const int rowBase = blockM + wm * 64 + i * 16 + mhalf;
#pragma unroll
            for (int r = 0; r < 8; ++r) {
                pre[(size_t)(rowBase + r) * NF + col] = acc[i][j][r] + bn;
            }
        }
    }
}

// ---------------------------------------------------------------------------
// 2) Top-k per token: iterative max with cached per-thread local maxima.
//    Ties broken by lowest index (matches jax.lax.top_k). relu on store.
//    Mutates `pre` (scratch; zeroed by the next kernel).
// ---------------------------------------------------------------------------
__global__ __launch_bounds__(256) void topk_kernel(
    float* __restrict__ pre, float* __restrict__ tv, int* __restrict__ ti)
{
    __shared__ float rv[256];
    __shared__ int   ri[256];

    const int t   = blockIdx.x;
    const int tid = threadIdx.x;
    float* row = pre + (size_t)t * NF;

    // thread owns strided elements tid + 256*i, i in [0,128)
    float bv = -__builtin_inff();
    int   bi = tid;
    for (int i = 0; i < 128; ++i) {
        int   idx = tid + (i << 8);
        float v   = row[idx];
        if (v > bv) { bv = v; bi = idx; }
    }

    for (int s = 0; s < KTOP; ++s) {
        rv[tid] = bv; ri[tid] = bi;
        __syncthreads();
        for (int off = 128; off > 0; off >>= 1) {
            if (tid < off) {
                float v2 = rv[tid + off]; int i2 = ri[tid + off];
                if (v2 > rv[tid] || (v2 == rv[tid] && i2 < ri[tid])) {
                    rv[tid] = v2; ri[tid] = i2;
                }
            }
            __syncthreads();
        }
        const float wv = rv[0];
        const int   wi = ri[0];
        if (tid == 0) {
            tv[t * KTOP + s] = fmaxf(wv, 0.0f);   // relu
            ti[t * KTOP + s] = wi;
        }
        __syncthreads();  // everyone read rv[0]/ri[0] before next overwrite
        if ((wi & 255) == tid) {                  // unique owner rescans
            row[wi] = -__builtin_inff();
            bv = -__builtin_inff(); bi = tid;
            for (int i = 0; i < 128; ++i) {
                int   idx = tid + (i << 8);
                float v   = row[idx];
                if (v > bv) { bv = v; bi = idx; }
            }
        }
        __syncthreads();
    }
}

// ---------------------------------------------------------------------------
// 3) zero features, 4) scatter relu'd top-k values
// ---------------------------------------------------------------------------
__global__ void zero_kernel(float4* __restrict__ p, int n4) {
    const int stride = gridDim.x * blockDim.x;
    for (int i = blockIdx.x * blockDim.x + threadIdx.x; i < n4; i += stride)
        p[i] = make_float4(0.f, 0.f, 0.f, 0.f);
}

__global__ void scatter_kernel(const float* __restrict__ tv,
                               const int* __restrict__ ti,
                               float* __restrict__ feat) {
    const int g = blockIdx.x * blockDim.x + threadIdx.x;   // [0, 4096*64)
    const int t = g >> 6;
    feat[(size_t)t * NF + ti[g]] = tv[g];
}

// ---------------------------------------------------------------------------
// 5) optional dec_w transpose: [D_OUT, NF] -> [NF, D_OUT] (coalesced gathers)
// ---------------------------------------------------------------------------
__global__ __launch_bounds__(256) void transpose_kernel(
    const float* __restrict__ src, float* __restrict__ dst)
{
    __shared__ float tile[32][33];
    const int c0 = blockIdx.x * 32;   // NF dim
    const int r0 = blockIdx.y * 32;   // D_OUT dim
    const int tx = threadIdx.x & 31;
    const int ty = threadIdx.x >> 5;  // 0..7
#pragma unroll
    for (int i = 0; i < 32; i += 8)
        tile[ty + i][tx] = src[(size_t)(r0 + ty + i) * NF + c0 + tx];
    __syncthreads();
#pragma unroll
    for (int i = 0; i < 32; i += 8)
        dst[(size_t)(c0 + ty + i) * T_DIM + r0 + tx] = tile[tx][ty + i];
}

// ---------------------------------------------------------------------------
// 6) sparse decode + fused squared-error partial (one WG per token).
//    dec element (d, c) lives at dec[c*cStride + d*dStride].
// ---------------------------------------------------------------------------
__global__ __launch_bounds__(256) void decode_kernel(
    const float* __restrict__ tv, const int* __restrict__ ti,
    const float* __restrict__ dec, long cStride, long dStride,
    const float* __restrict__ target,
    float* __restrict__ pred, float* __restrict__ partial)
{
    __shared__ float sv[KTOP];
    __shared__ int   si[KTOP];
    __shared__ float red[256];

    const int t   = blockIdx.x;
    const int tid = threadIdx.x;
    if (tid < KTOP) { sv[tid] = tv[t * KTOP + tid]; si[tid] = ti[t * KTOP + tid]; }
    __syncthreads();

    float acc[16];
#pragma unroll
    for (int i = 0; i < 16; ++i) acc[i] = 0.0f;

    for (int j = 0; j < KTOP; ++j) {
        const float v = sv[j];
        const long  c = si[j];
        const float* col = dec + c * cStride;
#pragma unroll
        for (int i = 0; i < 16; ++i)
            acc[i] += v * col[(long)(tid + (i << 8)) * dStride];
    }

    float lsum = 0.0f;
#pragma unroll
    for (int i = 0; i < 16; ++i) {
        const int d = tid + (i << 8);
        const float p = acc[i];
        pred[(size_t)t * T_DIM + d] = p;
        const float diff = p - target[(size_t)t * T_DIM + d];
        lsum += diff * diff;
    }

    red[tid] = lsum;
    __syncthreads();
    for (int off = 128; off > 0; off >>= 1) {
        if (tid < off) red[tid] += red[tid + off];
        __syncthreads();
    }
    if (tid == 0) partial[t] = red[0];
}

// ---------------------------------------------------------------------------
// 7) deterministic final reduction -> loss scalars
// ---------------------------------------------------------------------------
__global__ __launch_bounds__(256) void finalize_kernel(
    const float* __restrict__ partial, float* __restrict__ out_scalars)
{
    __shared__ float red[256];
    const int tid = threadIdx.x;
    float s = 0.0f;
    for (int i = tid; i < T_DIM; i += 256) s += partial[i];
    red[tid] = s;
    __syncthreads();
    for (int off = 128; off > 0; off >>= 1) {
        if (tid < off) red[tid] += red[tid + off];
        __syncthreads();
    }
    if (tid == 0) {
        const float pl = red[0] / 16777216.0f;   // mean over 4096*4096
        out_scalars[0] = pl;    // loss
        out_scalars[1] = pl;    // prediction_loss
        out_scalars[2] = 0.0f;  // sparsity_loss
    }
}

// ---------------------------------------------------------------------------
extern "C" void kernel_launch(void* const* d_in, const int* in_sizes, int n_in,
                              void* d_out, int out_size, void* d_ws, size_t ws_size,
                              hipStream_t stream) {
    const float* mlp_input    = (const float*)d_in[0];   // [4096, 4096]
    const float* mlp_output   = (const float*)d_in[1];   // [4096, 4096]
    const float* enc_w        = (const float*)d_in[2];   // [32768, 4096]
    const float* enc_b        = (const float*)d_in[3];   // [32768]
    const float* dec_w        = (const float*)d_in[4];   // [4096, 32768]
    const float* encoder_bias = (const float*)d_in[5];   // [4096]
    // d_in[6] = k (compile-time fixed to 64 per setup_inputs)

    float* out      = (float*)d_out;
    float* features = out;                                   // 134217728 floats
    float* pred     = out + 134217728ull;                    //  16777216 floats
    float* scalars  = out + 150994944ull;                    //  loss, pl, sl

    char*  ws      = (char*)d_ws;
    float* tv      = (float*)(ws);                           // 4096*64 f32 = 1 MB
    int*   ti      = (int*)(ws + (1ull << 20));              // 4096*64 i32 = 1 MB
    float* partial = (float*)(ws + (2ull << 20));            // 4096 f32
    float* decT    = (float*)(ws + (4ull << 20));            // 512 MB (optional)
    const bool useT = ws_size >= (4ull << 20) + 536870912ull;

    // 1) pre-activations (staged in the features region of d_out)
    encode_gemm_kernel<<<dim3(NF / BN, T_DIM / BM), 512, 0, stream>>>(
        mlp_input, enc_w, encoder_bias, enc_b, features);

    // 2) top-k per token (reads + mutates the staged pre)
    topk_kernel<<<T_DIM, 256, 0, stream>>>(features, tv, ti);

    // 3) zero features, 4) scatter relu'd top-k values
    zero_kernel<<<8192, 256, 0, stream>>>((float4*)features, 33554432);
    scatter_kernel<<<(T_DIM * KTOP) / 256, 256, 0, stream>>>(tv, ti, features);

    // 5/6) sparse decode (+ fused loss partials)
    const float* dec = dec_w;
    long cS = 1, dS = NF;                    // dec_w[d, c] at d*NF + c
    if (useT) {
        transpose_kernel<<<dim3(NF / 32, T_DIM / 32), 256, 0, stream>>>(dec_w, decT);
        dec = decT; cS = T_DIM; dS = 1;      // decT[c, d] at c*4096 + d
    }
    decode_kernel<<<T_DIM, 256, 0, stream>>>(tv, ti, dec, cS, dS,
                                             mlp_output, pred, partial);

    // 7) deterministic loss reduction
    finalize_kernel<<<1, 256, 0, stream>>>(partial, scalars);
}
